// MacroContextAdderToSubASTNodes_2705829396960
// MI455X (gfx1250) — matrix operand order repaired and checked by reading, not compiled, not run on hardware
//
#include <hip/hip_runtime.h>
#include <hip/hip_fp16.h>
#include <stdint.h>

#if defined(__gfx1250__) && __has_builtin(__builtin_amdgcn_global_load_async_to_lds_b128)
#define ASYNC_GATHER 1
#else
#define ASYNC_GATHER 0
#endif

typedef __attribute__((ext_vector_type(16))) _Float16 v16h;
typedef __attribute__((ext_vector_type(8)))  float    v8f;
typedef __attribute__((ext_vector_type(4)))  int      i32x4;

#define D_AST 256
#define D_CFG 256
#define K_GATE 512
#define TM 32                       // edges per block (two 16-row WMMA tiles)
#define LDS_STRIDE (D_AST + 8)      // 264 halves = 528 B, 16B-multiple, bank-conflict pad

#define GLOBAL_AS __attribute__((address_space(1)))
#define LDS_AS    __attribute__((address_space(3)))

// ---------------------------------------------------------------------------
// Pack a row-major f32 weight matrix W[K][N] (N==256) into f16 WMMA B-fragment
// order for v_wmma_f32_16x16x32_f16:
//   fragment f = kstep*(N/16)+ntile holds 32 lanes x 16 halves;
//   lane l: column n = ntile*16 + (l&15); halves h=0..15 -> K = kstep*32 + (l>>4)*16 + h
// ---------------------------------------------------------------------------
__global__ void pack_w_kernel(const float* __restrict__ W, _Float16* __restrict__ P,
                              int K, int N) {
    int idx = blockIdx.x * blockDim.x + threadIdx.x;
    if (idx >= K * N) return;
    int h      = idx & 15;
    int lane   = (idx >> 4) & 31;
    int t      = idx >> 9;              // kstep*ntiles + ntile
    int ntiles = N >> 4;
    int kstep  = t / ntiles;
    int ntile  = t - kstep * ntiles;
    int n = ntile * 16 + (lane & 15);
    int k = kstep * 32 + ((lane >> 4) << 4) + h;
    P[idx] = (_Float16)W[k * N + n];
}

struct F4x2 { float4 a, b; };

// A-fragment (16x32 f16, M x K) from padded f16 LDS tile (base already offset
// to the 16-row tile). lane l: row m = l&15, kg = l>>4;
// halves 0..15 = K kstep*32 + kg*16 .. +15 (contiguous, matches packed B)
__device__ __forceinline__ v16h load_a_frag(const _Float16* __restrict__ tile,
                                            int kstep, int lane) {
    int m  = lane & 15;
    int kg = lane >> 4;
    const _Float16* p = tile + m * LDS_STRIDE + kstep * 32 + kg * 16;
    F4x2 t;
    t.a = *reinterpret_cast<const float4*>(p);
    t.b = *reinterpret_cast<const float4*>(p + 8);
    return __builtin_bit_cast(v16h, t);
}

__device__ __forceinline__ v8f wmma_f16(v16h a, v16h b, v8f c) {
    return __builtin_amdgcn_wmma_f32_16x16x32_f16(false, a, false, b, (short)0, c,
                                                  false, false);
}

// 16-byte global -> LDS copy; async (ASYNCcnt) on gfx1250 toolchains that
// expose the builtin, plain VGPR bounce otherwise.
__device__ __forceinline__ void copy16_g2lds(const float* __restrict__ g,
                                             float* __restrict__ l) {
#if ASYNC_GATHER
    __builtin_amdgcn_global_load_async_to_lds_b128(
        (GLOBAL_AS i32x4*)g, (LDS_AS i32x4*)l, 0, 0);
#else
    *reinterpret_cast<float4*>(l) = *reinterpret_cast<const float4*>(g);
#endif
}

__device__ __forceinline__ void wait_gather() {
#if ASYNC_GATHER
#if __has_builtin(__builtin_amdgcn_s_wait_asynccnt)
    __builtin_amdgcn_s_wait_asynccnt(0);
#else
    asm volatile("s_wait_asynccnt 0" ::: "memory");
#endif
#endif
}

__launch_bounds__(256, 2)
__global__ void edge_gate_kernel(const float* __restrict__ prev,
                                 const float* __restrict__ cfg,
                                 const int*   __restrict__ kidx,
                                 const int*   __restrict__ vidx,
                                 const float* __restrict__ b_proj,
                                 const float* __restrict__ b_gate,
                                 const _Float16* __restrict__ Wp,  // packed f16 frags, K=256
                                 const _Float16* __restrict__ Wg,  // packed f16 frags, K=512
                                 float* __restrict__ out, int E) {
    __shared__ alignas(16) float    stS[TM * D_AST];        // staged s (f32), 32 KB
    __shared__ alignas(16) float    stU[TM * D_AST];        // staged u_raw (f32), 32 KB
    __shared__ alignas(16) _Float16 sS[TM * LDS_STRIDE];    // s (f16 tile)
    __shared__ alignas(16) _Float16 sU[TM * LDS_STRIDE];    // u_raw, then relu(u) (f16)
    __shared__ int rowK[TM];
    __shared__ int rowV[TM];

    const int tid  = threadIdx.x;
    const int lane = tid & 31;
    const int w    = tid >> 5;          // wave 0..7, owns columns [32w, 32w+32)
    const int e0   = blockIdx.x * TM;

    if (tid < TM) {
        int e = e0 + tid;
        if (e >= E) e = E - 1;          // clamp; WMMA needs full EXEC anyway
        rowK[tid] = kidx[e];
        rowV[tid] = vidx[e];
    }
    __syncthreads();

    // ---- gather s and u_raw rows into f32 LDS staging (async path) --------
    #pragma unroll
    for (int i = 0; i < (TM * (D_AST / 4)) / 256; ++i) {    // 8 x 16B per thread
        int id = tid + i * 256;
        int m  = id >> 6;               // 64 float4 per row
        int c4 = (id & 63) * 4;
        copy16_g2lds(prev + (size_t)rowK[m] * D_AST + c4, stS + m * D_AST + c4);
        copy16_g2lds(cfg  + (size_t)rowV[m] * D_CFG + c4, stU + m * D_AST + c4);
    }
    wait_gather();
    __syncthreads();

    // ---- f32 -> f16 conversion pass into padded WMMA tiles -----------------
    #pragma unroll
    for (int i = 0; i < (TM * (D_AST / 4)) / 256; ++i) {
        int id = tid + i * 256;
        int m  = id >> 6;
        int c4 = (id & 63) * 4;
        float4 vs = *reinterpret_cast<const float4*>(stS + m * D_AST + c4);
        float4 vu = *reinterpret_cast<const float4*>(stU + m * D_AST + c4);
        _Float16* ps = sS + m * LDS_STRIDE + c4;
        _Float16* pu = sU + m * LDS_STRIDE + c4;
        ps[0] = (_Float16)vs.x; ps[1] = (_Float16)vs.y;
        ps[2] = (_Float16)vs.z; ps[3] = (_Float16)vs.w;
        pu[0] = (_Float16)vu.x; pu[1] = (_Float16)vu.y;
        pu[2] = (_Float16)vu.z; pu[3] = (_Float16)vu.w;
    }
    __syncthreads();

    const int l15 = lane & 15;
    const int kg  = lane >> 4;
    const int n0  = w * 32 + l15;       // j=0 column
    const int n1  = n0 + 16;            // j=1 column
    const int ntA = 2 * w;
    const int ntB = 2 * w + 1;

    // ---- GEMM1: u = relu(u_raw @ W_proj + b_proj), K=256, M-tiles x2 -------
    v8f uacc[2][2];
    {
        float bp0 = b_proj[n0], bp1 = b_proj[n1];
        #pragma unroll
        for (int r = 0; r < 8; ++r) {
            uacc[0][0][r] = bp0; uacc[0][1][r] = bp1;
            uacc[1][0][r] = bp0; uacc[1][1][r] = bp1;
        }
    }
    #pragma unroll
    for (int ks = 0; ks < D_CFG / 32; ++ks) {
        v16h b0 = *reinterpret_cast<const v16h*>(Wp + ((ks * 16 + ntA) * 512) + lane * 16);
        v16h b1 = *reinterpret_cast<const v16h*>(Wp + ((ks * 16 + ntB) * 512) + lane * 16);
        #pragma unroll
        for (int mt = 0; mt < 2; ++mt) {            // B frags reused across 2 row tiles
            v16h a = load_a_frag(sU + mt * 16 * LDS_STRIDE, ks, lane);
            uacc[mt][0] = wmma_f16(a, b0, uacc[mt][0]);
            uacc[mt][1] = wmma_f16(a, b1, uacc[mt][1]);
        }
    }
    __syncthreads();                    // all waves done reading u_raw tile

    // relu; publish f16 u into sU (A-matrix cols 256..511 of GEMM2, epilogue u)
    #pragma unroll
    for (int mt = 0; mt < 2; ++mt) {
        #pragma unroll
        for (int r = 0; r < 8; ++r) {
            int m = mt * 16 + r + 8 * kg;
            float a0 = uacc[mt][0][r] > 0.0f ? uacc[mt][0][r] : 0.0f;
            float a1 = uacc[mt][1][r] > 0.0f ? uacc[mt][1][r] : 0.0f;
            sU[m * LDS_STRIDE + n0] = (_Float16)a0;
            sU[m * LDS_STRIDE + n1] = (_Float16)a1;
        }
    }
    __syncthreads();

    // ---- GEMM2: g = sigmoid([s,u] @ W_gate + b_gate), K=512 ----------------
    v8f gacc[2][2];
    {
        float bg0 = b_gate[n0], bg1 = b_gate[n1];
        #pragma unroll
        for (int r = 0; r < 8; ++r) {
            gacc[0][0][r] = bg0; gacc[0][1][r] = bg1;
            gacc[1][0][r] = bg0; gacc[1][1][r] = bg1;
        }
    }
    #pragma unroll
    for (int ks = 0; ks < K_GATE / 32; ++ks) {
        const _Float16* src = (ks < 8) ? sS : sU;   // cols 0..255 = s, 256..511 = u
        v16h b0 = *reinterpret_cast<const v16h*>(Wg + ((ks * 16 + ntA) * 512) + lane * 16);
        v16h b1 = *reinterpret_cast<const v16h*>(Wg + ((ks * 16 + ntB) * 512) + lane * 16);
        #pragma unroll
        for (int mt = 0; mt < 2; ++mt) {
            v16h a = load_a_frag(src + mt * 16 * LDS_STRIDE, ks & 7, lane);
            gacc[mt][0] = wmma_f16(a, b0, gacc[mt][0]);
            gacc[mt][1] = wmma_f16(a, b1, gacc[mt][1]);
        }
    }

    // ---- upd = g*s + (1-g)*u, scatter into out (s f32 from LDS stage) ------
    #pragma unroll
    for (int mt = 0; mt < 2; ++mt) {
        #pragma unroll
        for (int r = 0; r < 8; ++r) {
            int m = mt * 16 + r + 8 * kg;
            int e = e0 + m;
            if (e < E) {
                size_t row = (size_t)rowK[m] * D_AST;
                float s0 = stS[m * D_AST + n0];
                float s1 = stS[m * D_AST + n1];
                float u0 = (float)sU[m * LDS_STRIDE + n0];
                float u1 = (float)sU[m * LDS_STRIDE + n1];
                float gg0 = 1.0f / (1.0f + __expf(-gacc[mt][0][r]));
                float gg1 = 1.0f / (1.0f + __expf(-gacc[mt][1][r]));
                out[row + n0] = gg0 * s0 + (1.0f - gg0) * u0;
                out[row + n1] = gg1 * s1 + (1.0f - gg1) * u1;
            }
        }
    }
}

extern "C" void kernel_launch(void* const* d_in, const int* in_sizes, int n_in,
                              void* d_out, int out_size, void* d_ws, size_t ws_size,
                              hipStream_t stream) {
    const float* prev   = (const float*)d_in[0];
    const float* cfg    = (const float*)d_in[1];
    const int*   kidx   = (const int*)d_in[2];
    const int*   vidx   = (const int*)d_in[3];
    const float* W_proj = (const float*)d_in[4];
    const float* b_proj = (const float*)d_in[5];
    const float* W_gate = (const float*)d_in[6];
    const float* b_gate = (const float*)d_in[7];
    float* out = (float*)d_out;
    const int E = in_sizes[2];

    _Float16* wp = (_Float16*)d_ws;                 // 256*256 halves = 128 KB
    _Float16* wg = wp + D_CFG * D_AST;              // 512*256 halves = 256 KB

    // out starts as a copy of prev; scatter overwrites gathered rows.
    (void)hipMemcpyAsync(out, prev, (size_t)out_size * sizeof(float),
                         hipMemcpyDeviceToDevice, stream);

    pack_w_kernel<<<(D_CFG * D_AST + 255) / 256, 256, 0, stream>>>(W_proj, wp, D_CFG, D_AST);
    pack_w_kernel<<<(K_GATE * D_AST + 255) / 256, 256, 0, stream>>>(W_gate, wg, K_GATE, D_AST);

    edge_gate_kernel<<<(E + TM - 1) / TM, 256, 0, stream>>>(
        prev, cfg, kidx, vidx, b_proj, b_gate, wp, wg, out, E);
}
